// BasicAttention_3633542333230
// MI455X (gfx1250) — matrix operand order repaired and checked by reading, not compiled, not run on hardware
//
#include <hip/hip_runtime.h>
#include <hip/hip_bf16.h>

typedef __attribute__((ext_vector_type(16))) __bf16 v16bf;
typedef __attribute__((ext_vector_type(8)))  float  v8f;

#define BATCH 16
#define NK 2048
#define NV 2048
#define DK 1024
#define DV 1024
#define NWAVES 8
#define MTILE 32

// Workspace layout (fast path): values_bf16 [16*2048*1024] @ 0 (64 MB),
//                               W_bf16 [1024*1024] @ 64 MB (2 MB)
#define VAL_BF16_ELEMS ((size_t)BATCH * NV * DV)
#define W_BF16_ELEMS   ((size_t)DV * DK)
#define WS_NEEDED      ((VAL_BF16_ELEMS + W_BF16_ELEMS) * 2)

// LDS layout (bytes):
//   proj  bf16[32][1024]                    @ 0        (64 KB)
//   sc    16-bit [32][2048] f16 logits,     @ 65536    (128 KB)
//         overwritten in place by bf16 P
//   vt    bf16[1024][32]  V^T chunk         @ 196608   (64 KB)
//   rowsum f32[32]                          @ 262144
#define OFF_PROJ 0
#define OFF_SC   (64 * 1024)
#define OFF_VT   (192 * 1024)
#define OFF_RS   (256 * 1024)
#define SMEM_BYTES (256 * 1024 + 128)

__device__ __forceinline__ v16bf cvt16(float4 a, float4 b, float4 c, float4 d) {
  v16bf v;
  v[0]  = (__bf16)a.x; v[1]  = (__bf16)a.y; v[2]  = (__bf16)a.z; v[3]  = (__bf16)a.w;
  v[4]  = (__bf16)b.x; v[5]  = (__bf16)b.y; v[6]  = (__bf16)b.z; v[7]  = (__bf16)b.w;
  v[8]  = (__bf16)c.x; v[9]  = (__bf16)c.y; v[10] = (__bf16)c.z; v[11] = (__bf16)c.w;
  v[12] = (__bf16)d.x; v[13] = (__bf16)d.y; v[14] = (__bf16)d.z; v[15] = (__bf16)d.w;
  return v;
}

// Load 16 bf16 (two 16-byte runs) into a fragment (works for LDS or global).
__device__ __forceinline__ v16bf ld16bf(const __bf16* p0, const __bf16* p1) {
  union { uint4 q; __bf16 h[8]; } u0, u1;
  u0.q = *(const uint4*)p0;
  u1.q = *(const uint4*)p1;
  v16bf v;
#pragma unroll
  for (int i = 0; i < 8; ++i) { v[i] = u0.h[i]; v[i + 8] = u1.h[i]; }
  return v;
}

__device__ __forceinline__ unsigned short f32_to_f16_bits(float x) {
  _Float16 h = (_Float16)x;
  unsigned short u; __builtin_memcpy(&u, &h, 2); return u;
}
__device__ __forceinline__ float f16_bits_to_f32(unsigned short u) {
  _Float16 h; __builtin_memcpy(&h, &u, 2); return (float)h;
}
__device__ __forceinline__ unsigned short f32_to_bf16_bits(float x) {
  __bf16 h = (__bf16)x;
  unsigned short u; __builtin_memcpy(&u, &h, 2); return u;
}

// ---- Prep: convert an f32 array to bf16 (8 elements per thread) ----------------
__global__ __launch_bounds__(256)
void cvt_bf16_kernel(const float* __restrict__ src, __bf16* __restrict__ dst,
                     unsigned long long n) {
  unsigned long long i =
      ((unsigned long long)blockIdx.x * blockDim.x + threadIdx.x) * 8ull;
  if (i >= n) return;
  float4 a = *(const float4*)(src + i);
  float4 b = *(const float4*)(src + i + 4);
  union { uint4 q; __bf16 h[8]; } u;
  u.h[0] = (__bf16)a.x; u.h[1] = (__bf16)a.y; u.h[2] = (__bf16)a.z; u.h[3] = (__bf16)a.w;
  u.h[4] = (__bf16)b.x; u.h[5] = (__bf16)b.y; u.h[6] = (__bf16)b.z; u.h[7] = (__bf16)b.w;
  *(uint4*)(dst + i) = u.q;
}

template <bool PRE_BF16>
__global__ __launch_bounds__(256, 1)
void attn_fused_kernel(const float* __restrict__ values,
                       const int*   __restrict__ vmask,
                       const float* __restrict__ keys,
                       const float* __restrict__ W,
                       const __bf16* __restrict__ vbf,   // bf16 values (fast path)
                       const __bf16* __restrict__ wbf,   // bf16 W      (fast path)
                       float* __restrict__ out) {
  extern __shared__ char smem[];
  __bf16*         proj = (__bf16*)(smem + OFF_PROJ);          // [32][DV]
  unsigned short* sc   = (unsigned short*)(smem + OFF_SC);    // [32][NV] f16 -> bf16
  __bf16*         vt   = (__bf16*)(smem + OFF_VT);            // [1024][32]
  float*          rowsum = (float*)(smem + OFF_RS);           // [32]

  const int b    = blockIdx.x / (NK / MTILE);
  const int m0   = (blockIdx.x % (NK / MTILE)) * MTILE;
  const int tid  = threadIdx.x;
  const int wave = tid >> 5;
  const int lane = tid & 31;
  const int lhalf = lane >> 4;   // 0: lanes 0-15, 1: lanes 16-31
  const int lrow  = lane & 15;

  const float*  keysB  = keys + ((size_t)b * NK + m0) * DK;
  const float*  valB   = values + (size_t)b * NV * DV;
  const __bf16* valB16 = PRE_BF16 ? (vbf + (size_t)b * NV * DV) : nullptr;

  // ------- Phase 1: proj(32 x DV) = keys_tile(32 x DK) @ W^T, bf16 WMMA ----------
  {
    v8f acc[2][8];
#pragma unroll
    for (int h = 0; h < 2; ++h)
#pragma unroll
      for (int t = 0; t < 8; ++t) acc[h][t] = v8f{};

    for (int k0 = 0; k0 < DK; k0 += 32) {
      v16bf av[2];
#pragma unroll
      for (int h = 0; h < 2; ++h) {
        const float* ar = keysB + (size_t)(h * 16 + lrow) * DK + k0 + lhalf * 16;
        float4 a0 = *(const float4*)(ar + 0);
        float4 a1 = *(const float4*)(ar + 4);
        float4 a2 = *(const float4*)(ar + 8);
        float4 a3 = *(const float4*)(ar + 12);
        av[h] = cvt16(a0, a1, a2, a3);
      }
#pragma unroll
      for (int t = 0; t < 8; ++t) {
        const int n = wave * 128 + t * 16 + lrow;   // B column = W row (contiguous)
        v16bf bv;
        if constexpr (PRE_BF16) {
          const __bf16* br = wbf + (size_t)n * DK + k0 + lhalf * 16;
          bv = ld16bf(br, br + 8);
        } else {
          const float* br = W + (size_t)n * DK + k0 + lhalf * 16;
          bv = cvt16(((const float4*)br)[0], ((const float4*)br)[1],
                     ((const float4*)br)[2], ((const float4*)br)[3]);
        }
        acc[0][t] = __builtin_amdgcn_wmma_f32_16x16x32_bf16(
            false, av[0], false, bv, (short)0, acc[0][t], false, false);
        acc[1][t] = __builtin_amdgcn_wmma_f32_16x16x32_bf16(
            false, av[1], false, bv, (short)0, acc[1][t], false, false);
      }
    }
#pragma unroll
    for (int h = 0; h < 2; ++h)
#pragma unroll
      for (int t = 0; t < 8; ++t) {
        const int n = wave * 128 + t * 16 + lrow;
#pragma unroll
        for (int r = 0; r < 8; ++r)
          proj[(size_t)(h * 16 + r + lhalf * 8) * DV + n] = (__bf16)acc[h][t][r];
      }
  }
  __syncthreads();

  // ------- Phase 2: logits(32 x NV) = proj @ values^T into LDS (f16) -------------
  {
    const int n_base = wave * (NV / NWAVES);   // 256 columns per wave
    for (int t = 0; t < 16; ++t) {
      const int n0 = n_base + t * 16;
      v8f a0 = v8f{}, a1 = v8f{};
      for (int k0 = 0; k0 < DV; k0 += 32) {
        const __bf16* ap0 = proj + (size_t)lrow * DV + k0 + lhalf * 16;
        const __bf16* ap1 = proj + (size_t)(16 + lrow) * DV + k0 + lhalf * 16;
        v16bf av0 = ld16bf(ap0, ap0 + 8);
        v16bf av1 = ld16bf(ap1, ap1 + 8);
        v16bf bv;
        if constexpr (PRE_BF16) {
          const __bf16* br = valB16 + (size_t)(n0 + lrow) * DV + k0 + lhalf * 16;
          bv = ld16bf(br, br + 8);
        } else {
          const float* br = valB + (size_t)(n0 + lrow) * DV + k0 + lhalf * 16;
          bv = cvt16(((const float4*)br)[0], ((const float4*)br)[1],
                     ((const float4*)br)[2], ((const float4*)br)[3]);
        }
        a0 = __builtin_amdgcn_wmma_f32_16x16x32_bf16(
            false, av0, false, bv, (short)0, a0, false, false);
        a1 = __builtin_amdgcn_wmma_f32_16x16x32_bf16(
            false, av1, false, bv, (short)0, a1, false, false);
      }
#pragma unroll
      for (int r = 0; r < 8; ++r) {
        sc[(size_t)(r + lhalf * 8) * NV + n0 + lrow]      = f32_to_f16_bits(a0[r]);
        sc[(size_t)(16 + r + lhalf * 8) * NV + n0 + lrow] = f32_to_f16_bits(a1[r]);
      }
    }
  }
  __syncthreads();

  // ------- Phase 3: masked softmax over NV; P written back in place as bf16 -------
  {
    const int hw = wave * 2 + lhalf;   // 16 half-waves, each does 2 rows
    const int* mrow = vmask + (size_t)b * NV;
#pragma unroll
    for (int rr = 0; rr < 2; ++rr) {
      const int row = hw + rr * 16;
      unsigned short* srow = sc + (size_t)row * NV;
      float mx = -3.0e38f;
      for (int j = 0; j < NV / 16; ++j) {
        const int c = lrow + j * 16;
        float x = f16_bits_to_f32(srow[c]);
        if (mrow[c] == 0) x = -3.0e38f;
        srow[c] = f32_to_f16_bits(x);   // -inf in f16 if masked
        mx = fmaxf(mx, x);
      }
#pragma unroll
      for (int o = 8; o > 0; o >>= 1) mx = fmaxf(mx, __shfl_xor(mx, o, 16));
      float sum = 0.f;
      for (int j = 0; j < NV / 16; ++j) {
        const int c = lrow + j * 16;
        float p = __expf(f16_bits_to_f32(srow[c]) - mx);
        srow[c] = f32_to_bf16_bits(p);  // unnormalized P, bf16, in place
        sum += p;
      }
#pragma unroll
      for (int o = 8; o > 0; o >>= 1) sum += __shfl_xor(sum, o, 16);
      if (lrow == 0) rowsum[row] = sum;
    }
  }
  __syncthreads();

  // ------- Phase 4: out(32 x DV) = P(32 x NV) @ values, staged V^T in LDS ---------
  {
    v8f acc[2][8];
#pragma unroll
    for (int h = 0; h < 2; ++h)
#pragma unroll
      for (int t = 0; t < 8; ++t) acc[h][t] = v8f{};

    const __bf16* scp = (const __bf16*)sc;

    for (int n0 = 0; n0 < NV; n0 += 32) {
      // Cooperative transpose-stage: values[b, n0:n0+32, :] -> vt[d][n] bf16
      {
        const int p   = tid & 15;        // n-pair index: rows n0+2p, n0+2p+1
        const int dch = tid >> 4;        // 16 chunks of 64 d-columns
        unsigned* wbase = (unsigned*)vt;
        if constexpr (PRE_BF16) {
          const __bf16* r0 = valB16 + (size_t)(n0 + 2 * p) * DV + dch * 64;
          const __bf16* r1 = r0 + DV;
          for (int d = 0; d < 64; d += 8) {
            union { uint4 q; unsigned short s[8]; } q0, q1;
            q0.q = *(const uint4*)(r0 + d);
            q1.q = *(const uint4*)(r1 + d);
#pragma unroll
            for (int j = 0; j < 8; ++j)
              wbase[(size_t)(dch * 64 + d + j) * 16 + p] =
                  (unsigned)q0.s[j] | ((unsigned)q1.s[j] << 16);
          }
        } else {
          const float* r0 = valB + (size_t)(n0 + 2 * p) * DV + dch * 64;
          const float* r1 = r0 + DV;
          for (int d = 0; d < 64; d += 4) {
            float4 x0 = *(const float4*)(r0 + d);
            float4 x1 = *(const float4*)(r1 + d);
            union { unsigned u; __bf16 h[2]; } pk;
#pragma unroll
            for (int q = 0; q < 4; ++q) {
              const float e0 = (q == 0) ? x0.x : (q == 1) ? x0.y : (q == 2) ? x0.z : x0.w;
              const float e1 = (q == 0) ? x1.x : (q == 1) ? x1.y : (q == 2) ? x1.z : x1.w;
              pk.h[0] = (__bf16)e0;
              pk.h[1] = (__bf16)e1;
              wbase[(size_t)(dch * 64 + d + q) * 16 + p] = pk.u;
            }
          }
        }
      }
      __syncthreads();

      // A fragments straight from bf16 P in LDS (no conversion)
      const __bf16* ap0 = scp + (size_t)lrow * NV + n0 + lhalf * 16;
      const __bf16* ap1 = scp + (size_t)(16 + lrow) * NV + n0 + lhalf * 16;
      v16bf av0 = ld16bf(ap0, ap0 + 8);
      v16bf av1 = ld16bf(ap1, ap1 + 8);
#pragma unroll
      for (int t = 0; t < 8; ++t) {
        const int d0 = wave * 128 + t * 16;
        const __bf16* bp = vt + (size_t)(d0 + lrow) * 32 + lhalf * 16;
        v16bf bv = ld16bf(bp, bp + 8);
        acc[0][t] = __builtin_amdgcn_wmma_f32_16x16x32_bf16(
            false, av0, false, bv, (short)0, acc[0][t], false, false);
        acc[1][t] = __builtin_amdgcn_wmma_f32_16x16x32_bf16(
            false, av1, false, bv, (short)0, acc[1][t], false, false);
      }
      __syncthreads();   // protect vt before next staging round
    }

    // Epilogue: normalize by row sum and store fp32 output
    float inv[2][8];
#pragma unroll
    for (int h = 0; h < 2; ++h)
#pragma unroll
      for (int r = 0; r < 8; ++r)
        inv[h][r] = 1.0f / rowsum[h * 16 + r + lhalf * 8];
    float* outB = out + ((size_t)b * NK + m0) * DV;
#pragma unroll
    for (int h = 0; h < 2; ++h)
#pragma unroll
      for (int t = 0; t < 8; ++t) {
        const int d0 = wave * 128 + t * 16;
#pragma unroll
        for (int r = 0; r < 8; ++r)
          outB[(size_t)(h * 16 + r + lhalf * 8) * DV + d0 + lrow] =
              acc[h][t][r] * inv[h][r];
      }
  }
}

extern "C" void kernel_launch(void* const* d_in, const int* in_sizes, int n_in,
                              void* d_out, int out_size, void* d_ws, size_t ws_size,
                              hipStream_t stream) {
  (void)in_sizes; (void)n_in; (void)out_size;
  const float* values = (const float*)d_in[0];
  const int*   vmask  = (const int*)d_in[1];
  const float* keys   = (const float*)d_in[2];
  const float* W      = (const float*)d_in[3];
  float* out = (float*)d_out;

  static bool attr_set = false;
  if (!attr_set) {
    (void)hipFuncSetAttribute((const void*)attn_fused_kernel<true>,
                              hipFuncAttributeMaxDynamicSharedMemorySize,
                              SMEM_BYTES);
    (void)hipFuncSetAttribute((const void*)attn_fused_kernel<false>,
                              hipFuncAttributeMaxDynamicSharedMemorySize,
                              SMEM_BYTES);
    attr_set = true;
  }

  dim3 grid(BATCH * (NK / MTILE));
  dim3 block(256);

  if (ws_size >= WS_NEEDED) {
    __bf16* vbf = (__bf16*)d_ws;
    __bf16* wbf = vbf + VAL_BF16_ELEMS;
    // Convert values and W to bf16 once (pure-bandwidth prep)
    {
      unsigned long long nv = VAL_BF16_ELEMS;
      unsigned nb = (unsigned)((nv / 8 + 255) / 256);
      cvt_bf16_kernel<<<nb, 256, 0, stream>>>(values, vbf, nv);
      unsigned long long nw = W_BF16_ELEMS;
      unsigned nb2 = (unsigned)((nw / 8 + 255) / 256);
      cvt_bf16_kernel<<<nb2, 256, 0, stream>>>(W, wbf, nw);
    }
    attn_fused_kernel<true><<<grid, block, SMEM_BYTES, stream>>>(
        values, vmask, keys, W, vbf, wbf, out);
  } else {
    attn_fused_kernel<false><<<grid, block, SMEM_BYTES, stream>>>(
        values, vmask, keys, W, nullptr, nullptr, out);
  }
}
